// LSTMModel_2834678415370
// MI455X (gfx1250) — compile-verified, matrix operand order checked by reading, not run on hardware
//
#include <hip/hip_runtime.h>

// ---------------------------------------------------------------------------
// Problem dimensions (fixed by the reference)
// ---------------------------------------------------------------------------
#define TT   16384      // timesteps
#define II   1024       // input size
#define HH   1024       // hidden size
#define FH   4096       // 4*H gate rows

typedef __attribute__((ext_vector_type(16))) __bf16 v16bf;
typedef __attribute__((ext_vector_type(8)))  float  v8f;
typedef unsigned int u32x4 __attribute__((ext_vector_type(4)));
typedef int          i32x4 __attribute__((ext_vector_type(4)));
typedef int          i32x8 __attribute__((ext_vector_type(8)));

// address-space-qualified int4 for the async-LDS builtin
typedef __attribute__((address_space(1))) i32x4 as1_i32x4;
typedef __attribute__((address_space(3))) i32x4 as3_i32x4;

#ifndef __has_builtin
#define __has_builtin(x) 0
#endif

#if __has_builtin(__builtin_amdgcn_global_load_async_to_lds_b128)
#define HAVE_ASYNC 1
#else
#define HAVE_ASYNC 0
#endif
#if __has_builtin(__builtin_amdgcn_tensor_load_to_lds)
#define HAVE_TDM 1
#else
#define HAVE_TDM 0
#endif

__device__ __forceinline__ void wait_async0() {
#if __has_builtin(__builtin_amdgcn_s_wait_asynccnt)
    __builtin_amdgcn_s_wait_asynccnt(0);
#else
    asm volatile("s_wait_asynccnt 0x0" ::: "memory");
#endif
}

__device__ __forceinline__ void wait_tensor0() {
#if __has_builtin(__builtin_amdgcn_s_wait_tensorcnt)
    __builtin_amdgcn_s_wait_tensorcnt(0);
#else
    asm volatile("s_wait_tensorcnt 0x0" ::: "memory");
#endif
}

#if HAVE_ASYNC
// Direct global -> LDS async copy (16B per lane), ASYNCcnt-tracked.
__device__ __forceinline__ void async_b128(const void* g, void* l) {
    __builtin_amdgcn_global_load_async_to_lds_b128(
        (as1_i32x4*)(uintptr_t)g,
        (as3_i32x4*)(unsigned)(uintptr_t)l,
        0, 0);
}
#endif

// ---------------------------------------------------------------------------
// fp32 -> bf16 conversion (grid-stride)
// ---------------------------------------------------------------------------
__global__ void f32_to_bf16_kernel(const float* __restrict__ in,
                                   unsigned short* __restrict__ out,
                                   size_t n) {
    size_t i = (size_t)blockIdx.x * blockDim.x + threadIdx.x;
    size_t stride = (size_t)gridDim.x * blockDim.x;
    for (; i < n; i += stride) {
        __bf16 b = (__bf16)in[i];
        out[i] = __builtin_bit_cast(unsigned short, b);
    }
}

// ---------------------------------------------------------------------------
// Zero-init of h double-buffer + grid-barrier counter (fresh every call).
// ---------------------------------------------------------------------------
__global__ void init_state_kernel(float* __restrict__ hbuf,
                                  unsigned* __restrict__ syncc) {
    int tid = threadIdx.x;
    for (int i = tid; i < 2 * HH; i += blockDim.x) hbuf[i] = 0.0f;
    if (tid == 0) syncc[0] = 0u;
}

// ---------------------------------------------------------------------------
// WMMA fragment gather (ISA 7.12.2 16-bit A/B 16x32 layout). The 16 bf16 a
// lane needs are two contiguous 16-byte chunks: shorts [m*32+8*kh, +8) and
// [m*32+16+8*kh, +8)  ->  exactly two ds_load_b128.
// ---------------------------------------------------------------------------
__device__ __forceinline__ v16bf load_frag16x32(const unsigned short* __restrict__ base,
                                                int lane) {
    const int m  = lane & 15;
    const int kh = lane >> 4;
    union { v16bf v; uint4 q[2]; } f;
    const uint4* p = reinterpret_cast<const uint4*>(base + m * 32 + kh * 8);
    f.q[0] = p[0];   // K pairs {0,1,2,3,4,5,6,7} (+8*kh)
    f.q[1] = p[2];   // K pairs {16..23} (+8*kh)  : +32 bytes
    return f.v;
}

#define WMMA_BF16(A, B, C) \
    __builtin_amdgcn_wmma_f32_16x16x32_bf16(false, (A), false, (B), (short)0, (C), false, false)

// ---------------------------------------------------------------------------
// Phase 1: x_gates = X @ W_ih^T + (b_ih+b_hh). bf16 WMMA, f32 accumulate.
// 128x128 tile per WG; K in 32-chunks; double-buffered LDS fed by
// GLOBAL_LOAD_ASYNC_TO_LDS_B128 (one barrier per K-step) when available.
// Named accumulators + all B-frags loaded before the 8 back-to-back WMMAs
// (avoids accumulator shuffling / WMMA->VALU hazard NOPs).
// ---------------------------------------------------------------------------
__global__ void __launch_bounds__(256)
gates_gemm_kernel(const unsigned short* __restrict__ Abf,   // [T][I] bf16
                  const unsigned short* __restrict__ Wbf,   // [4H][I] bf16
                  const float* __restrict__ b_ih,
                  const float* __restrict__ b_hh,
                  float* __restrict__ xg) {                  // [T][4H]
    __shared__ alignas(16) unsigned short sA[2][128 * 32];
    __shared__ alignas(16) unsigned short sB[2][128 * 32];

    const int tid   = threadIdx.x;
    const int lane  = tid & 31;
    const int wv    = tid >> 5;            // wave 0..7
    const int mbase = blockIdx.y * 128;
    const int nbase = blockIdx.x * 128;

    // cooperative staging indices: 128 rows x 32 bf16, 16 elems per thread
    const int cr = tid >> 1;               // row 0..127
    const int c0 = (tid & 1) * 16;         // col 0 or 16

    const v8f vz = (v8f){0.f, 0.f, 0.f, 0.f, 0.f, 0.f, 0.f, 0.f};
    v8f d0 = vz, d1 = vz, d2 = vz, d3 = vz, d4 = vz, d5 = vz, d6 = vz, d7 = vz;

    const int NK = II / 32;

#if HAVE_ASYNC
#define STAGE(kc, buf) do {                                                 \
        const size_t ga_ = (size_t)(mbase + cr) * II + (kc) * 32 + c0;      \
        async_b128(Abf + ga_,     &sA[buf][cr * 32 + c0]);                  \
        async_b128(Abf + ga_ + 8, &sA[buf][cr * 32 + c0 + 8]);              \
        const size_t gb_ = (size_t)(nbase + cr) * II + (kc) * 32 + c0;      \
        async_b128(Wbf + gb_,     &sB[buf][cr * 32 + c0]);                  \
        async_b128(Wbf + gb_ + 8, &sB[buf][cr * 32 + c0 + 8]);              \
    } while (0)

    STAGE(0, 0);
    wait_async0();
    __syncthreads();
    for (int kc = 0; kc < NK; ++kc) {
        const int buf = kc & 1;
        if (kc + 1 < NK) STAGE(kc + 1, buf ^ 1);   // overlap with WMMA below

        const unsigned short* pB = &sB[buf][0];
        const v16bf a  = load_frag16x32(&sA[buf][(wv * 16) * 32], lane);
        const v16bf b0 = load_frag16x32(pB + 0 * 512, lane);
        const v16bf b1 = load_frag16x32(pB + 1 * 512, lane);
        const v16bf b2 = load_frag16x32(pB + 2 * 512, lane);
        const v16bf b3 = load_frag16x32(pB + 3 * 512, lane);
        const v16bf b4 = load_frag16x32(pB + 4 * 512, lane);
        const v16bf b5 = load_frag16x32(pB + 5 * 512, lane);
        const v16bf b6 = load_frag16x32(pB + 6 * 512, lane);
        const v16bf b7 = load_frag16x32(pB + 7 * 512, lane);

        d0 = WMMA_BF16(a, b0, d0);
        d1 = WMMA_BF16(a, b1, d1);
        d2 = WMMA_BF16(a, b2, d2);
        d3 = WMMA_BF16(a, b3, d3);
        d4 = WMMA_BF16(a, b4, d4);
        d5 = WMMA_BF16(a, b5, d5);
        d6 = WMMA_BF16(a, b6, d6);
        d7 = WMMA_BF16(a, b7, d7);

        wait_async0();
        __syncthreads();
    }
#undef STAGE
#else
    for (int kc = 0; kc < NK; ++kc) {
        const uint4* ga = reinterpret_cast<const uint4*>(
            Abf + ((size_t)(mbase + cr) * II + kc * 32 + c0));
        uint4* qa = reinterpret_cast<uint4*>(&sA[0][cr * 32 + c0]);
        qa[0] = ga[0]; qa[1] = ga[1];
        const uint4* gb = reinterpret_cast<const uint4*>(
            Wbf + ((size_t)(nbase + cr) * II + kc * 32 + c0));
        uint4* qb = reinterpret_cast<uint4*>(&sB[0][cr * 32 + c0]);
        qb[0] = gb[0]; qb[1] = gb[1];
        __syncthreads();
        const unsigned short* pB = &sB[0][0];
        const v16bf a  = load_frag16x32(&sA[0][(wv * 16) * 32], lane);
        const v16bf b0 = load_frag16x32(pB + 0 * 512, lane);
        const v16bf b1 = load_frag16x32(pB + 1 * 512, lane);
        const v16bf b2 = load_frag16x32(pB + 2 * 512, lane);
        const v16bf b3 = load_frag16x32(pB + 3 * 512, lane);
        const v16bf b4 = load_frag16x32(pB + 4 * 512, lane);
        const v16bf b5 = load_frag16x32(pB + 5 * 512, lane);
        const v16bf b6 = load_frag16x32(pB + 6 * 512, lane);
        const v16bf b7 = load_frag16x32(pB + 7 * 512, lane);
        d0 = WMMA_BF16(a, b0, d0);
        d1 = WMMA_BF16(a, b1, d1);
        d2 = WMMA_BF16(a, b2, d2);
        d3 = WMMA_BF16(a, b3, d3);
        d4 = WMMA_BF16(a, b4, d4);
        d5 = WMMA_BF16(a, b5, d5);
        d6 = WMMA_BF16(a, b6, d6);
        d7 = WMMA_BF16(a, b7, d7);
        __syncthreads();
    }
#endif

    // epilogue: C layout — VGPR r, lanes 0-15: M=r, lanes 16-31: M=r+8
    const int mrow0 = mbase + wv * 16 + ((lane >> 4) * 8);
#define EPI(s, dd) do {                                                     \
        const int n_ = nbase + (s) * 16 + (lane & 15);                      \
        const float bias_ = b_ih[n_] + b_hh[n_];                            \
        _Pragma("unroll")                                                   \
        for (int r = 0; r < 8; ++r)                                         \
            xg[(size_t)(mrow0 + r) * FH + n_] = (dd)[r] + bias_;            \
    } while (0)
    EPI(0, d0); EPI(1, d1); EPI(2, d2); EPI(3, d3);
    EPI(4, d4); EPI(5, d5); EPI(6, d6); EPI(7, d7);
#undef EPI
}

// ---------------------------------------------------------------------------
// TDM: load 16 consecutive W_hh rows (16 x 1024 f32, row stride 1024) into
// LDS, pad_enable inserting 1 DWORD after every 256 DWORDs -> each row
// occupies 4x257 = 1028 DWORDs (the bank-conflict-free layout the dot loop
// wants). D# per ISA ch.8. Issued by wave 0 only; TENSORcnt-tracked.
// ---------------------------------------------------------------------------
#if HAVE_TDM
__device__ __forceinline__ void tdm_load_16rows(const float* gsrc, void* lds_dst) {
    const unsigned long long ga = (unsigned long long)(uintptr_t)gsrc;
    u32x4 g0;
    g0[0] = 1u;                                        // count=1 (valid), user mode
    g0[1] = (unsigned)(uintptr_t)lds_dst;              // lds_addr (bytes)
    g0[2] = (unsigned)(ga & 0xffffffffu);              // global_addr[31:0]
    g0[3] = (unsigned)((ga >> 32) & 0x01ffffffu)       // global_addr[56:32]
            | (2u << 30);                              // type=2 ("image")
    i32x8 g1;
    g1[0] = (2 << 16)       // data_size = 4B
          | (1 << 20)       // pad_enable
          | (7 << 22)       // pad_interval: 256 DWORDs
          | (0 << 25);      // pad_amount: 1 DWORD
    g1[1] = (int)(1024u << 16);   // tensor_dim0 = 1024
    g1[2] = (int)(16u << 16);     // tensor_dim1 = 16
    g1[3] = (int)(1024u << 16);   // tile_dim0 = 1024
    g1[4] = 16;                   // tile_dim1 = 16 rows
    g1[5] = 1024;                 // tensor_dim0_stride = 1024 elems
    g1[6] = 0;
    g1[7] = 0;
    const i32x4 z4 = (i32x4){0, 0, 0, 0};
#if defined(__clang_major__) && __clang_major__ >= 23
    const i32x8 z8 = (i32x8){0, 0, 0, 0, 0, 0, 0, 0};
    __builtin_amdgcn_tensor_load_to_lds(g0, g1, z4, z4, z8, 0);
#else
    __builtin_amdgcn_tensor_load_to_lds(g0, g1, z4, z4, 0);
#endif
}
#endif

// ---------------------------------------------------------------------------
// Phase 2: persistent LSTM recurrence. 64 WGs x 256 thr; WG w owns h-lanes
// [w*16,w*16+16) and W_hh rows {g*1024+w*16+j}. Padded fp32 slice resident
// in LDS; (4*row+q+k) mod 64 covers all banks -> conflict-free dot loop.
// One monotonic L2-atomic grid barrier per timestep.
// ---------------------------------------------------------------------------
#define GRID2 64
#define WROW  1028                  // padded row stride (dwords)
#define QOFF  257                   // padded quarter-block stride (dwords)

__global__ void __launch_bounds__(256)
lstm_recurrence_kernel(const float* __restrict__ xg,    // [T][4096]
                       const float* __restrict__ Whh,   // [4096][1024]
                       float* __restrict__ hbuf,        // [2][1024]
                       unsigned* __restrict__ syncc) {
    extern __shared__ float smem[];
    float* sW = smem;                       // [64][1028] padded weight slice
    float* sh = smem + 64 * WROW;           // [4*257]    padded h
    float* sy = sh + 4 * QOFF;              // [64]       gate pre-activations

    const int tid   = threadIdx.x;
    const int wg    = blockIdx.x;           // 0..63
    const int hbase = wg * 16;

    // --- one-time W_hh slice preload into the padded LDS layout ------------
#if HAVE_TDM
    if (tid < 32) {                         // one wave issues 4 TDM descriptors
#pragma unroll
        for (int g = 0; g < 4; ++g) {
            const float* gsrc = Whh + (size_t)(g * HH + hbase) * HH;
            tdm_load_16rows(gsrc, (char*)sW + (size_t)g * 16 * WROW * 4);
        }
        wait_tensor0();
    }
#else
    for (int l = 0; l < 64; ++l) {
        const int grow = (l >> 4) * HH + hbase + (l & 15);
        const float* src = Whh + (size_t)grow * HH;
        for (int c = tid; c < HH; c += 256)
            sW[l * WROW + (c >> 8) * QOFF + (c & 255)] = src[c];
    }
#endif
    float c_state = 0.0f;                   // valid for tid < 16
    const int row = tid >> 2;               // 0..63 : output row in slice
    const int q   = tid & 3;                // 0..3  : k-quarter
    __syncthreads();

    for (int t = 0; t < TT; ++t) {
        // h(t) into padded LDS (4*tid..4*tid+3 stay inside one 256-block)
        {
            const float4 hv = reinterpret_cast<const float4*>(hbuf + (t & 1) * HH)[tid];
            const int e = 4 * tid;
            float* d = sh + (e >> 8) * QOFF + (e & 255);
            d[0] = hv.x; d[1] = hv.y; d[2] = hv.z; d[3] = hv.w;
        }
        __syncthreads();

        // warm next timestep's gate rows (global_prefetch_b8)
        if (tid < 64 && t + 1 < TT) {
            const float* nxt = xg + (size_t)(t + 1) * FH
                               + (tid >> 4) * HH + hbase + (tid & 15);
            __builtin_prefetch(nxt, 0, 1);
        }

        // 64 dot products of length 1024; 4 lanes per row, conflict-free LDS
        const float* wr = sW + row * WROW + q * QOFF;
        const float* hv = sh + q * QOFF;
        float acc = 0.0f;
#pragma unroll 16
        for (int k = 0; k < 256; ++k) acc = __builtin_fmaf(wr[k], hv[k], acc);
        acc += __shfl_xor(acc, 1, 32);      // partials sit in adjacent lanes
        acc += __shfl_xor(acc, 2, 32);
        if (q == 0) sy[row] = acc;
        __syncthreads();

        if (tid < 16) {
            const float* xgt = xg + (size_t)t * FH + hbase + tid;
            const float yi = sy[tid]      + xgt[0];
            const float yf = sy[16 + tid] + xgt[HH];
            const float yg = sy[32 + tid] + xgt[2 * HH];
            const float yo = sy[48 + tid] + xgt[3 * HH];
            const float ig = 1.0f / (1.0f + __expf(-yi));
            const float fg = 1.0f / (1.0f + __expf(-yf));
            const float gg = tanhf(yg);
            const float og = 1.0f / (1.0f + __expf(-yo));
            c_state = fg * c_state + ig * gg;
            hbuf[((t + 1) & 1) * HH + hbase + tid] = og * tanhf(c_state);
        }

        // grid-wide release/acquire barrier (monotonic counter)
        __threadfence();
        __syncthreads();
        if (tid == 0) {
            __hip_atomic_fetch_add(syncc, 1u, __ATOMIC_RELEASE,
                                   __HIP_MEMORY_SCOPE_AGENT);
            const unsigned target = (unsigned)(t + 1) * GRID2;
            while (__hip_atomic_load(syncc, __ATOMIC_ACQUIRE,
                                     __HIP_MEMORY_SCOPE_AGENT) < target)
                __builtin_amdgcn_s_sleep(1);
        }
        __syncthreads();
    }
}

// ---------------------------------------------------------------------------
// Phase 3: out[0] = h_last . W_lin + b_lin   (h_last = hbuf[0..1023])
// ---------------------------------------------------------------------------
__global__ void final_proj_kernel(const float* __restrict__ h,
                                  const float* __restrict__ Wlin,
                                  const float* __restrict__ blin,
                                  float* __restrict__ out) {
    __shared__ float red[256];
    const int tid = threadIdx.x;
    float a = 0.0f;
    for (int k = tid; k < HH; k += 256) a = __builtin_fmaf(h[k], Wlin[k], a);
    red[tid] = a;
    __syncthreads();
    for (int s = 128; s > 0; s >>= 1) {
        if (tid < s) red[tid] += red[tid + s];
        __syncthreads();
    }
    if (tid == 0) out[0] = red[0] + blin[0];
}

// ---------------------------------------------------------------------------
// Launch: ws = [x_gates f32 256MB][X bf16 32MB][W_ih bf16 8MB][h][sync at h+4KB]
// ---------------------------------------------------------------------------
extern "C" void kernel_launch(void* const* d_in, const int* in_sizes, int n_in,
                              void* d_out, int out_size, void* d_ws, size_t ws_size,
                              hipStream_t stream) {
    const float* x    = (const float*)d_in[0];
    const float* Wih  = (const float*)d_in[1];
    const float* Whh  = (const float*)d_in[2];
    const float* bih  = (const float*)d_in[3];
    const float* bhh  = (const float*)d_in[4];
    const float* Wlin = (const float*)d_in[5];
    const float* blin = (const float*)d_in[6];
    float* out = (float*)d_out;

    char* ws = (char*)d_ws;
    const size_t XG_BYTES = (size_t)TT * FH * sizeof(float);       // 256 MB
    const size_t A_BYTES  = (size_t)TT * II * 2;                   // 32 MB
    const size_t W_BYTES  = (size_t)FH * II * 2;                   // 8 MB
    float*          xgates = (float*)ws;
    unsigned short* Abf    = (unsigned short*)(ws + XG_BYTES);
    unsigned short* Wbf    = (unsigned short*)(ws + XG_BYTES + A_BYTES);
    float*          hbuf   = (float*)(ws + XG_BYTES + A_BYTES + W_BYTES);
    unsigned*       syncc  = (unsigned*)(ws + XG_BYTES + A_BYTES + W_BYTES + 4096);

    f32_to_bf16_kernel<<<4096, 256, 0, stream>>>(x,   Abf, (size_t)TT * II);
    f32_to_bf16_kernel<<<2048, 256, 0, stream>>>(Wih, Wbf, (size_t)FH * II);
    init_state_kernel<<<1, 256, 0, stream>>>(hbuf, syncc);

    gates_gemm_kernel<<<dim3(FH / 128, TT / 128), 256, 0, stream>>>(
        Abf, Wbf, bih, bhh, xgates);

    const size_t shmem = (size_t)(64 * WROW + 4 * QOFF + 64) * sizeof(float);
    (void)hipFuncSetAttribute((const void*)lstm_recurrence_kernel,
                              hipFuncAttributeMaxDynamicSharedMemorySize,
                              (int)shmem);
    lstm_recurrence_kernel<<<GRID2, 256, shmem, stream>>>(xgates, Whh, hbuf, syncc);

    final_proj_kernel<<<1, 256, 0, stream>>>(hbuf, Wlin, blin, out);
}
